// TransformerFrustumEncoder_3358664425621
// MI455X (gfx1250) — matrix operand order repaired and checked by reading, not compiled, not run on hardware
//
#include <hip/hip_runtime.h>
#include <hip/hip_fp16.h>
#include <math.h>
#include <stdint.h>

typedef __attribute__((ext_vector_type(16))) _Float16 v16h;
typedef __attribute__((ext_vector_type(8)))  float    v8f;

#define NPOL 64
#define NAZ  256
#define NSY  (NPOL*NAZ)   // 16384 pseudo-image rows
#define NH   4            // heads
#define ND   32           // channels per head
#define NE   128          // embedding = NH*ND
#define NCP  128          // projection channels
#define NL   2            // encoder layers
#define NCIN 8
#define NF   8192         // nonempty frustums
#define NP   16           // points per frustum

// ---------------------------------------------------------------------------
// WMMA helpers (CDNA5 wave32, V_WMMA_F32_16X16X32_F16)
// ---------------------------------------------------------------------------
__device__ __forceinline__ v8f wmma16(v16h a, v16h b, v8f c) {
  return __builtin_amdgcn_wmma_f32_16x16x32_f16(
      /*neg_a=*/false, a, /*neg_b=*/false, b,
      /*c_mod=*/(short)0, c, /*reuse_a=*/false, /*reuse_b=*/false);
}

__device__ __forceinline__ float gelu_exact(float x) {
  return 0.5f * x * (1.0f + erff(x * 0.70710678118654752f));
}

// A fragment (16x32 f16) from an f32 LDS tile; rows m=0..15, cols kbase..kbase+31.
// Layout: lanes 0-15 hold K 0-15; lanes 16-31 hold K 16-31 (matches B fragments).
// k >= kmax is zero-padded (used to pad attention K=16 up to 32).
__device__ __forceinline__ v16h fragA(const float* base, int stride, int kbase,
                                      int kmax, int lane) {
  int m = lane & 15;
  int koff = (lane & 16) ? 16 : 0;
  v16h a;
#pragma unroll
  for (int j = 0; j < 16; ++j) {
    int r = j >> 1, c = j & 1;
    int k = koff + 2 * r + c;
    a[j] = (k < kmax) ? (_Float16)base[m * stride + kbase + k] : (_Float16)0.0f;
  }
  return a;
}

// B fragment (32x16 f16): element (k,n) = base[(kbase+k)*stride + nbase + n].
// Layout: lanes 0-15 hold K 0-15 (N=lane); lanes 16-31 hold K 16-31 (N=lane-16).
__device__ __forceinline__ v16h fragB(const float* base, int stride, int kbase,
                                      int nbase, int kmax, int lane) {
  int n = lane & 15;
  int koff = (lane & 16) ? 16 : 0;
  v16h b;
#pragma unroll
  for (int j = 0; j < 16; ++j) {
    int k = koff + j;
    b[j] = (k < kmax) ? (_Float16)base[(kbase + k) * stride + nbase + n]
                      : (_Float16)0.0f;
  }
  return b;
}

// B fragment from a transposed source (for Q @ K^T): (k,n) = base[(nbase+n)*stride + kbase+k]
__device__ __forceinline__ v16h fragBT(const float* base, int stride, int kbase,
                                       int nbase, int lane) {
  int n = lane & 15;
  int koff = (lane & 16) ? 16 : 0;
  v16h b;
#pragma unroll
  for (int j = 0; j < 16; ++j) {
    int k = koff + j;
    b[j] = (_Float16)base[(nbase + n) * stride + kbase + k];
  }
  return b;
}

// Pre-swizzled weight fragment: one aligned 32-byte vector load per lane.
__device__ __forceinline__ v16h fragW(const _Float16* swz, int fragIdx, int lane) {
  return *(const v16h*)(swz + ((size_t)fragIdx * 32 + lane) * 16);
}

// ---------------------------------------------------------------------------
// Tensor Data Mover: stage a 2D f32 tile (tile_d1 rows x tile_d0 elems) from
// global memory into LDS. D# layout per CDNA5 ISA.
// g1w0 carries data_size (+ optional pad_enable/pad_interval/pad_amount).
// ---------------------------------------------------------------------------
#if defined(__gfx1250__) && __has_builtin(__builtin_amdgcn_tensor_load_to_lds)
#define HAVE_TDM 1
typedef unsigned int u32x4 __attribute__((ext_vector_type(4)));
typedef int i32x4 __attribute__((ext_vector_type(4)));
typedef int i32x8 __attribute__((ext_vector_type(8)));

__device__ __forceinline__ void tdm_load_tile_2d(unsigned lds_addr,
                                                 const void* gaddr,
                                                 unsigned tensor_d0,
                                                 unsigned tensor_d1,
                                                 unsigned tile_d0,
                                                 unsigned tile_d1,
                                                 unsigned d0_stride,
                                                 unsigned g1w0) {
  unsigned long long ga = (unsigned long long)(uintptr_t)gaddr;
  u32x4 g0;
  g0[0] = 1u;                                              // count=1 (valid user D#)
  g0[1] = lds_addr;                                        // LDS byte address
  g0[2] = (unsigned)(ga & 0xffffffffu);                    // global_addr[31:0]
  g0[3] = (unsigned)((ga >> 32) & 0x1ffffffu) | (2u << 30); // addr[56:32] | type=2
  i32x8 g1;
  g1[0] = (int)g1w0;                                       // mask=0 | data_size | pad cfg
  g1[1] = (int)((tensor_d0 & 0xffffu) << 16);              // tensor_dim0[15:0]
  g1[2] = (int)(((tensor_d0 >> 16) & 0xffffu) |
                ((tensor_d1 & 0xffffu) << 16));            // dim0 hi | dim1 lo
  g1[3] = (int)(((tensor_d1 >> 16) & 0xffffu) |
                ((tile_d0 & 0xffffu) << 16));              // dim1 hi | tile_dim0
  g1[4] = (int)(tile_d1 & 0xffffu);                        // tile_dim1 | tile_dim2=0
  g1[5] = (int)d0_stride;                                  // tensor_dim0_stride[31:0]
  g1[6] = 0;                                               // stride hi | dim1_stride lo
  g1[7] = 0;
  i32x4 gz = {0, 0, 0, 0};
#if __clang_major__ >= 23
  i32x8 gz8 = {0, 0, 0, 0, 0, 0, 0, 0};
  __builtin_amdgcn_tensor_load_to_lds(g0, g1, gz, gz, gz8, 0);
#else
  __builtin_amdgcn_tensor_load_to_lds(g0, g1, gz, gz, 0);
#endif
}
#define TDM_WAIT() __builtin_amdgcn_s_wait_tensorcnt((short)0)
#endif

// data_size=4B                       -> group1 word0
#define TDM_G1W0_B32        0x00020000u
// + pad_enable, pad_interval=128 DW (code 6), pad_amount=4 DW (code 3)
#define TDM_G1W0_B32_PAD132 0x07920000u

// ---------------------------------------------------------------------------
// Prep: f32 [K,N] weight -> f16 B-fragment-ordered buffer
// dst[((kc*NT+nt)*32 + lane)*16 + j] = src[k][n], lane=(kk>=16?16:0)+nn, j=kk&15
// ---------------------------------------------------------------------------
__global__ void swizzle_kernel(const float* __restrict__ src,
                               _Float16* __restrict__ dst, int K, int N) {
  int idx = blockIdx.x * blockDim.x + threadIdx.x;
  if (idx >= K * N) return;
  int k = idx / N, n = idx - k * N;
  int kc = k >> 5, kk = k & 31, nt = n >> 4, nn = n & 15;
  int NT = N >> 4;
  int lane = ((kk >= 16) ? 16 : 0) + nn;
  int j = kk & 15;
  dst[((size_t)((kc * NT + nt) * 32 + lane)) * 16 + j] = (_Float16)src[idx];
}

__global__ void zero_kernel(float* p, int n) {
  int i = blockIdx.x * blockDim.x + threadIdx.x;
  if (i < n) p[i] = 0.0f;
}

__global__ void scatter_kernel(const float* __restrict__ src,
                               const int* __restrict__ idx,
                               float* __restrict__ feats, int f, int c) {
  int i = blockIdx.x * blockDim.x + threadIdx.x;
  if (i >= f * c) return;
  int fi = i / c, ci = i - fi * c;
  feats[(size_t)idx[fi] * c + ci] = src[i];
}

// ---------------------------------------------------------------------------
// Fused per-frustum encoder: one wave32 per frustum.
// fc_pre -> 2x(LN, MHSA, LN, FFN) -> LN -> kv -> cross-attn -> outF[f][128]
// ---------------------------------------------------------------------------
__global__ __launch_bounds__(32) void encoder_kernel(
    const float* __restrict__ pc,
    const float* __restrict__ fc_pre_w, const float* __restrict__ fc_pre_b,
    const _Float16* __restrict__ swzQ, const float* __restrict__ qkv_b,
    const float* __restrict__ sa_g, const float* __restrict__ sa_b,
    const float* __restrict__ ffn_g, const float* __restrict__ ffn_bn,
    const _Float16* __restrict__ swzF1, const float* __restrict__ ffn_b1,
    const _Float16* __restrict__ swzF2, const float* __restrict__ ffn_b2,
    const float* __restrict__ ca_g, const float* __restrict__ ca_b,
    const _Float16* __restrict__ swzKV, const float* __restrict__ kv_b,
    const float* __restrict__ decoder, const int* __restrict__ i_unique,
    float* __restrict__ outF) {
  __shared__ float xs[NP][132];      // persistent activations [16 x 128]
  __shared__ float hs[NP][132];      // layernorm output (A operand source)
  __shared__ float ts[NP][264];      // gemm outputs: qkv / scores / ffn-hidden / kv
  __shared__ float ptile[NP * NCIN]; // TDM-staged pc tile (16 x 8, contiguous)

  const int f = blockIdx.x;
  const int lane = threadIdx.x;
  const int n = lane & 15;
  const int moff = (lane & 16) ? 8 : 0;

  // ---- stage the frustum's 16x8 point tile into LDS (TDM: one 512B row) ----
#ifdef HAVE_TDM
  tdm_load_tile_2d((unsigned)(uintptr_t)&ptile[0], pc + (size_t)f * NP * NCIN,
                   /*tensor_d0=*/NP * NCIN, /*tensor_d1=*/1,
                   /*tile_d0=*/NP * NCIN, /*tile_d1=*/1,
                   /*d0_stride=*/NP * NCIN, TDM_G1W0_B32);
  TDM_WAIT();
#else
  for (int i = lane; i < NP * NCIN; i += 32)
    ptile[i] = pc[(size_t)f * NP * NCIN + i];
#endif
  __syncthreads();

  // ---- fc_pre: [16 x 8] @ [8 x 128] + b (K=8: plain VALU) ----
  for (int p = 0; p < NP; ++p) {
    for (int c = lane; c < NE; c += 32) {
      float acc = fc_pre_b[c];
#pragma unroll
      for (int i = 0; i < NCIN; ++i)
        acc += ptile[p * NCIN + i] * fc_pre_w[i * NE + c];
      xs[p][c] = acc;
    }
  }
  __syncthreads();

  const float inv_sqrt_d = 0.17677669529663687f;  // 1/sqrt(32)

  for (int l = 0; l < NL; ++l) {
    // ---- LN over D=32 per (point, head): 64 groups, 2 per lane ----
#pragma unroll
    for (int t = 0; t < 2; ++t) {
      int grp = lane * 2 + t;
      int p = grp >> 2, h = grp & 3;
      const float* xrow = &xs[p][h * ND];
      float m = 0.0f;
      for (int d = 0; d < ND; ++d) m += xrow[d];
      m *= (1.0f / ND);
      float v = 0.0f;
      for (int d = 0; d < ND; ++d) { float dd = xrow[d] - m; v += dd * dd; }
      v *= (1.0f / ND);
      float inv = rsqrtf(v + 1e-5f);
      for (int d = 0; d < ND; ++d)
        hs[p][h * ND + d] = (xrow[d] - m) * inv * sa_g[l * ND + d] + sa_b[l * ND + d];
    }
    __syncthreads();

    const _Float16* wq = swzQ + (size_t)l * 6 * 512;
    for (int h = 0; h < NH; ++h) {
      // ---- qkv: [16x32] @ [32x96] -> ts cols 0..95 (q|k|v) ----
      v16h ah = fragA(&hs[0][0], 132, h * ND, 32, lane);
      for (int nt = 0; nt < 6; ++nt) {
        v8f c = {};
        c = wmma16(ah, fragW(wq, nt, lane), c);
        int col = nt * 16 + n;
#pragma unroll
        for (int r = 0; r < 8; ++r) ts[moff + r][col] = c[r] + qkv_b[l * 96 + col];
      }
      __syncthreads();

      // ---- scores = Q @ K^T * 1/sqrt(D) -> ts cols 96..111 ----
      {
        v16h aq = fragA(&ts[0][0], 264, 0, 32, lane);
        v16h bk = fragBT(&ts[0][0], 264, /*kbaseCol=*/ND, /*nbaseRow=*/0, lane);
        v8f sc = {};
        sc = wmma16(aq, bk, sc);
#pragma unroll
        for (int r = 0; r < 8; ++r) ts[moff + r][96 + n] = sc[r] * inv_sqrt_d;
      }
      __syncthreads();

      // ---- softmax over 16 keys (one row per lane 0..15) ----
      if (lane < 16) {
        float mx = -1e30f;
        for (int qy = 0; qy < NP; ++qy) mx = fmaxf(mx, ts[lane][96 + qy]);
        float s = 0.0f;
        for (int qy = 0; qy < NP; ++qy) {
          float e = expf(ts[lane][96 + qy] - mx);
          ts[lane][96 + qy] = e;
          s += e;
        }
        float inv = 1.0f / s;
        for (int qy = 0; qy < NP; ++qy) ts[lane][96 + qy] *= inv;
      }
      __syncthreads();

      // ---- attn @ V (K padded 16->32 with zeros) + residual into xs ----
      {
        v16h aa = fragA(&ts[0][0], 264, 96, /*kmax=*/16, lane);
#pragma unroll
        for (int nt = 0; nt < 2; ++nt) {
          v16h bv = fragB(&ts[0][0], 264, /*kbase=*/0, /*nbase=*/64 + nt * 16,
                          /*kmax=*/16, lane);
          v8f o = {};
          o = wmma16(aa, bv, o);
          int col = h * ND + nt * 16 + n;
#pragma unroll
          for (int r = 0; r < 8; ++r) xs[moff + r][col] += o[r];
        }
      }
      __syncthreads();
    }

    // ---- FFN LayerNorm ----
#pragma unroll
    for (int t = 0; t < 2; ++t) {
      int grp = lane * 2 + t;
      int p = grp >> 2, h = grp & 3;
      const float* xrow = &xs[p][h * ND];
      float m = 0.0f;
      for (int d = 0; d < ND; ++d) m += xrow[d];
      m *= (1.0f / ND);
      float v = 0.0f;
      for (int d = 0; d < ND; ++d) { float dd = xrow[d] - m; v += dd * dd; }
      v *= (1.0f / ND);
      float inv = rsqrtf(v + 1e-5f);
      for (int d = 0; d < ND; ++d)
        hs[p][h * ND + d] = (xrow[d] - m) * inv * ffn_g[l * ND + d] + ffn_bn[l * ND + d];
    }
    __syncthreads();

    const _Float16* w1 = swzF1 + (size_t)l * 8 * 512;
    const _Float16* w2 = swzF2 + (size_t)l * 8 * 512;
    for (int h = 0; h < NH; ++h) {
      // ---- hidden = gelu([16x32] @ [32x128] + b1) -> ts cols 0..127 ----
      v16h ah = fragA(&hs[0][0], 132, h * ND, 32, lane);
      for (int nt = 0; nt < 8; ++nt) {
        v8f c = {};
        c = wmma16(ah, fragW(w1, nt, lane), c);
        int col = nt * 16 + n;
#pragma unroll
        for (int r = 0; r < 8; ++r)
          ts[moff + r][col] = gelu_exact(c[r] + ffn_b1[l * 128 + col]);
      }
      __syncthreads();

      // ---- out = hidden[16x128] @ [128x32] + b2 + residual ----
      v16h af[4];
#pragma unroll
      for (int kc = 0; kc < 4; ++kc) af[kc] = fragA(&ts[0][0], 264, kc * 32, 32, lane);
#pragma unroll
      for (int nt = 0; nt < 2; ++nt) {
        v8f c = {};
#pragma unroll
        for (int kc = 0; kc < 4; ++kc)
          c = wmma16(af[kc], fragW(w2, kc * 2 + nt, lane), c);
        int col = h * ND + nt * 16 + n;
#pragma unroll
        for (int r = 0; r < 8; ++r)
          xs[moff + r][col] += c[r] + ffn_b2[l * ND + nt * 16 + n];
      }
      __syncthreads();
    }
  }

  // ---- cross-attn LayerNorm over full E=128 per point ----
  if (lane < 16) {
    const float* xrow = &xs[lane][0];
    float m = 0.0f;
    for (int d = 0; d < NE; ++d) m += xrow[d];
    m *= (1.0f / NE);
    float v = 0.0f;
    for (int d = 0; d < NE; ++d) { float dd = xrow[d] - m; v += dd * dd; }
    v *= (1.0f / NE);
    float inv = rsqrtf(v + 1e-5f);
    for (int d = 0; d < NE; ++d)
      hs[lane][d] = (xrow[d] - m) * inv * ca_g[d] + ca_b[d];
  }
  __syncthreads();

  // ---- kv = hn[16x128] @ [128x256] + b -> ts (k: 0..127, v: 128..255) ----
  {
    v16h ak[4];
#pragma unroll
    for (int kc = 0; kc < 4; ++kc) ak[kc] = fragA(&hs[0][0], 132, kc * 32, 32, lane);
    for (int nt = 0; nt < 16; ++nt) {
      v8f c = {};
#pragma unroll
      for (int kc = 0; kc < 4; ++kc)
        c = wmma16(ak[kc], fragW(swzKV, kc * 16 + nt, lane), c);
      int col = nt * 16 + n;
#pragma unroll
      for (int r = 0; r < 8; ++r) ts[moff + r][col] = c[r] + kv_b[col];
    }
  }
  __syncthreads();

  // ---- cross attention: one query per frustum ----
  const int iu = i_unique[f];
  const float* q = decoder + (size_t)(iu % NPOL) * NCP;
  if (lane < 16) {
    float s = 0.0f;
    for (int c = 0; c < NCP; ++c) s += q[c] * ts[lane][c];
    xs[0][lane] = s * 0.08838834764831845f;  // 1/sqrt(128)
  }
  __syncthreads();
  float mx = -1e30f;
  for (int j = 0; j < NP; ++j) mx = fmaxf(mx, xs[0][j]);
  float den = 0.0f;
  for (int j = 0; j < NP; ++j) den += expf(xs[0][j] - mx);
  if (lane < 16) xs[1][lane] = expf(xs[0][lane] - mx) / den;
  __syncthreads();
  for (int c = lane; c < NCP; c += 32) {
    float acc = q[c];
    for (int j = 0; j < NP; ++j) acc += xs[1][j] * ts[j][NCP + c];
    outF[(size_t)f * NCP + c] = acc;
  }
}

// ---------------------------------------------------------------------------
// Pseudo-image FFN: per 16-row tile: LN -> gelu(X@W1+b1)@W2 + b2 + X
// feats tile staged once via TDM (padded to 132-float row stride in LDS).
// ---------------------------------------------------------------------------
__global__ __launch_bounds__(32) void pseudo_ffn_kernel(
    const float* __restrict__ feats,
    const float* __restrict__ g, const float* __restrict__ b,
    const _Float16* __restrict__ w1s, const float* __restrict__ b1,
    const _Float16* __restrict__ w2s, const float* __restrict__ b2,
    float* __restrict__ out) {
  __shared__ float ft[16][132];   // TDM-staged feats tile (pad 128->132)
  __shared__ float hn[16][132];
  __shared__ float hid[16][520];
  const int r0 = blockIdx.x * 16;
  const int lane = threadIdx.x;
  const int n = lane & 15;
  const int moff = (lane & 16) ? 8 : 0;

  // light prefetch hint for the fragment-ordered weights (global_prefetch_b8)
  __builtin_prefetch(w1s + (size_t)lane * 512, 0, 1);
  __builtin_prefetch(w2s + (size_t)lane * 512, 0, 1);

#ifdef HAVE_TDM
  tdm_load_tile_2d((unsigned)(uintptr_t)&ft[0][0], feats + (size_t)r0 * NCP,
                   /*tensor_d0=*/NCP, /*tensor_d1=*/16,
                   /*tile_d0=*/NCP, /*tile_d1=*/16,
                   /*d0_stride=*/NCP, TDM_G1W0_B32_PAD132);
  TDM_WAIT();
#else
  for (int i = lane; i < 16 * NCP; i += 32)
    ft[i >> 7][i & 127] = feats[(size_t)r0 * NCP + i];
#endif
  __syncthreads();

  if (lane < 16) {
    const float* xrow = &ft[lane][0];
    float m = 0.0f;
    for (int d = 0; d < NCP; ++d) m += xrow[d];
    m *= (1.0f / NCP);
    float v = 0.0f;
    for (int d = 0; d < NCP; ++d) { float dd = xrow[d] - m; v += dd * dd; }
    v *= (1.0f / NCP);
    float inv = rsqrtf(v + 1e-5f);
    for (int d = 0; d < NCP; ++d) hn[lane][d] = (xrow[d] - m) * inv * g[d] + b[d];
  }
  __syncthreads();

  // hidden = gelu([16x128] @ [128x512] + b1)
  {
    v16h af[4];
#pragma unroll
    for (int kc = 0; kc < 4; ++kc) af[kc] = fragA(&hn[0][0], 132, kc * 32, 32, lane);
    for (int nt = 0; nt < 32; ++nt) {
      v8f c = {};
#pragma unroll
      for (int kc = 0; kc < 4; ++kc)
        c = wmma16(af[kc], fragW(w1s, kc * 32 + nt, lane), c);
      int col = nt * 16 + n;
#pragma unroll
      for (int r = 0; r < 8; ++r) hid[moff + r][col] = gelu_exact(c[r] + b1[col]);
    }
  }
  __syncthreads();

  // out = hidden[16x512] @ [512x128] + b2 + feats (k-blocked A reuse)
  {
    v8f acc[8];
#pragma unroll
    for (int nt = 0; nt < 8; ++nt) acc[nt] = (v8f){};
    for (int kb = 0; kb < 4; ++kb) {
      v16h a4[4];
#pragma unroll
      for (int j = 0; j < 4; ++j)
        a4[j] = fragA(&hid[0][0], 520, (kb * 4 + j) * 32, 32, lane);
#pragma unroll
      for (int nt = 0; nt < 8; ++nt) {
#pragma unroll
        for (int j = 0; j < 4; ++j)
          acc[nt] = wmma16(a4[j], fragW(w2s, (kb * 4 + j) * 8 + nt, lane), acc[nt]);
      }
    }
#pragma unroll
    for (int nt = 0; nt < 8; ++nt) {
      int col = nt * 16 + n;
#pragma unroll
      for (int r = 0; r < 8; ++r) {
        size_t row = (size_t)(r0 + moff + r);
        out[row * NCP + col] = acc[nt][r] + b2[col] + ft[moff + r][col];
      }
    }
  }
}

// ---------------------------------------------------------------------------
extern "C" void kernel_launch(void* const* d_in, const int* in_sizes, int n_in,
                              void* d_out, int out_size, void* d_ws, size_t ws_size,
                              hipStream_t stream) {
  (void)in_sizes; (void)n_in; (void)out_size; (void)ws_size;
  const float* pc       = (const float*)d_in[0];
  const float* fc_pre_w = (const float*)d_in[1];
  const float* fc_pre_b = (const float*)d_in[2];
  const float* qkv_w    = (const float*)d_in[3];
  const float* qkv_b    = (const float*)d_in[4];
  const float* sa_g     = (const float*)d_in[5];
  const float* sa_b     = (const float*)d_in[6];
  const float* ffn_g    = (const float*)d_in[7];
  const float* ffn_bn   = (const float*)d_in[8];
  const float* ffn_w1   = (const float*)d_in[9];
  const float* ffn_b1   = (const float*)d_in[10];
  const float* ffn_w2   = (const float*)d_in[11];
  const float* ffn_b2   = (const float*)d_in[12];
  const float* ca_g     = (const float*)d_in[13];
  const float* ca_b     = (const float*)d_in[14];
  const float* kv_w     = (const float*)d_in[15];
  const float* kv_b     = (const float*)d_in[16];
  const float* decoder  = (const float*)d_in[17];
  const float* cff_g    = (const float*)d_in[18];
  const float* cff_b    = (const float*)d_in[19];
  const float* cff_w1   = (const float*)d_in[20];
  const float* cff_b1   = (const float*)d_in[21];
  const float* cff_w2   = (const float*)d_in[22];
  const float* cff_b2   = (const float*)d_in[23];
  const int*   i_unique = (const int*)d_in[24];
  float* out = (float*)d_out;

  char* ws = (char*)d_ws;
  size_t cur = 0;
  auto walloc = [&](size_t bytes) -> void* {
    void* p = ws + cur;
    cur = (cur + bytes + 255) & ~(size_t)255;
    return p;
  };
  _Float16* swzQ  = (_Float16*)walloc((size_t)NL * 6 * 512 * sizeof(_Float16));
  _Float16* swzF1 = (_Float16*)walloc((size_t)NL * 8 * 512 * sizeof(_Float16));
  _Float16* swzF2 = (_Float16*)walloc((size_t)NL * 8 * 512 * sizeof(_Float16));
  _Float16* swzKV = (_Float16*)walloc((size_t)4 * 16 * 512 * sizeof(_Float16));
  _Float16* swzC1 = (_Float16*)walloc((size_t)4 * 32 * 512 * sizeof(_Float16));
  _Float16* swzC2 = (_Float16*)walloc((size_t)16 * 8 * 512 * sizeof(_Float16));
  float*    outF  = (float*)walloc((size_t)NF * NCP * sizeof(float));
  float*    feats = (float*)walloc((size_t)NSY * NCP * sizeof(float));

#define SWZ(SRC, DST, K, N) \
  swizzle_kernel<<<((K) * (N) + 255) / 256, 256, 0, stream>>>((SRC), (DST), (K), (N))
  for (int l = 0; l < NL; ++l) {
    SWZ(qkv_w + (size_t)l * 32 * 96, swzQ + (size_t)l * 6 * 512, 32, 96);
    SWZ(ffn_w1 + (size_t)l * 32 * 128, swzF1 + (size_t)l * 8 * 512, 32, 128);
    SWZ(ffn_w2 + (size_t)l * 128 * 32, swzF2 + (size_t)l * 8 * 512, 128, 32);
  }
  SWZ(kv_w, swzKV, 128, 256);
  SWZ(cff_w1, swzC1, 128, 512);
  SWZ(cff_w2, swzC2, 512, 128);
#undef SWZ

  encoder_kernel<<<NF, 32, 0, stream>>>(
      pc, fc_pre_w, fc_pre_b, swzQ, qkv_b, sa_g, sa_b, ffn_g, ffn_bn,
      swzF1, ffn_b1, swzF2, ffn_b2, ca_g, ca_b, swzKV, kv_b, decoder,
      i_unique, outF);

  zero_kernel<<<(NSY * NCP + 255) / 256, 256, 0, stream>>>(feats, NSY * NCP);
  scatter_kernel<<<(NF * NCP + 255) / 256, 256, 0, stream>>>(outF, i_unique,
                                                             feats, NF, NCP);
  pseudo_ffn_kernel<<<NSY / 16, 32, 0, stream>>>(feats, cff_g, cff_b, swzC1,
                                                 cff_b1, swzC2, cff_b2, out);
}
